// TAG_40132174414164
// MI455X (gfx1250) — matrix operand order repaired and checked by reading, not compiled, not run on hardware
//
#include <hip/hip_runtime.h>
#include <hip/hip_bf16.h>
#include <stdint.h>

// ---------------- problem constants (from reference) ----------------
#define NN 50000
#define EE 800000
#define PP 65536
#define FIN 128      // feature width for all SpMM hops
#define KCAT 512     // (K+1)*128 concatenated features

// ---------------- vector types ----------------
typedef __attribute__((ext_vector_type(16))) __bf16         v16bf;
typedef __attribute__((ext_vector_type(16))) unsigned short ushort16;
typedef __attribute__((ext_vector_type(8)))  unsigned short ushort8;
typedef __attribute__((ext_vector_type(8)))  float          v8f;

struct APack { ushort8 lo, hi; };   // 32 bytes -> bit_cast to v16bf
struct Frags { APack a; ushort16 b0, b1, b2, b3; };

static __device__ __forceinline__ unsigned short f2bf(float f) {
    unsigned int u = __float_as_uint(f);
    u += 0x7FFFu + ((u >> 16) & 1u);   // round-to-nearest-even
    return (unsigned short)(u >> 16);
}

// ---------------- degree kernels ----------------
__global__ void k_zero(float* p, int n) {
    int t = blockIdx.x * blockDim.x + threadIdx.x;
    if (t < n) p[t] = 0.0f;
}
__global__ void k_deg_scatter(const int* __restrict__ dst, float* __restrict__ deg) {
    int e = blockIdx.x * blockDim.x + threadIdx.x;
    if (e < EE) atomicAdd(&deg[dst[e]], 1.0f);
}
__global__ void k_deg_finalize(float* deg) {
    int t = blockIdx.x * blockDim.x + threadIdx.x;
    if (t < NN) deg[t] = rsqrtf(fmaxf(deg[t], 1.0f));
}

// ---------------- concat-buffer helpers (row stride KCAT, f32) ----------------
__global__ void k_copy_slice0(const float* __restrict__ src, float* __restrict__ F4) {
    int t = blockIdx.x * blockDim.x + threadIdx.x;
    if (t >= NN * FIN) return;
    int row = t >> 7, col = t & 127;
    F4[(size_t)row * KCAT + col] = src[t];
}
__global__ void k_zero_slice(float* __restrict__ slice) {
    int t = blockIdx.x * blockDim.x + threadIdx.x;
    if (t >= NN * FIN) return;
    int row = t >> 7, col = t & 127;
    slice[(size_t)row * KCAT + col] = 0.0f;
}
__global__ void k_scale_slice(float* __restrict__ slice, const float* __restrict__ dn) {
    int t = blockIdx.x * blockDim.x + threadIdx.x;
    if (t >= NN * FIN) return;
    int row = t >> 7, col = t & 127;
    slice[(size_t)row * KCAT + col] *= dn[row];
}

// ---------------- SpMM hop: edge-parallel scatter-atomic (f32) ----------------
__global__ void k_edge_scatter(const int* __restrict__ src, const int* __restrict__ dst,
                               const float* __restrict__ w, const float* __restrict__ dn,
                               const float* __restrict__ prev, float* __restrict__ outS) {
    long long t = (long long)blockIdx.x * blockDim.x + threadIdx.x;
    if (t >= (long long)EE * 32) return;
    int e  = (int)(t >> 5);
    int ch = (int)(t & 31);
    int s = src[e], d = dst[e];
    float coef = w[e] * dn[s];
    const float4 v = *(const float4*)(prev + (size_t)s * KCAT + ch * 4);
    float* o = outS + (size_t)d * KCAT + ch * 4;
    atomicAdd(o + 0, v.x * coef);
    atomicAdd(o + 1, v.y * coef);
    atomicAdd(o + 2, v.z * coef);
    atomicAdd(o + 3, v.w * coef);
}

// ---------------- one-shot conversions for the WMMA GEMM ----------------
__global__ void k_cvt_feats(const float* __restrict__ F4, unsigned short* __restrict__ Fb) {
    long long t = (long long)blockIdx.x * blockDim.x + threadIdx.x;
    if (t >= (long long)NN * KCAT) return;
    Fb[t] = f2bf(F4[t]);
}
// W (512 x H f32, row-major) -> Wt (H x 512 bf16, transposed)
__global__ void k_cvt_wt(const float* __restrict__ W, unsigned short* __restrict__ Wt, int H) {
    int t = blockIdx.x * blockDim.x + threadIdx.x;
    if (t >= KCAT * H) return;
    int n = t / KCAT, k = t - n * KCAT;
    Wt[(size_t)n * KCAT + k] = f2bf(W[(size_t)k * H + n]);
}

// ---------------- WMMA GEMM: out = act(Fb(NxKCAT) @ Wt^T + b) ----------------
static __device__ __forceinline__ Frags load_frags(const unsigned short* __restrict__ arow,
                                                   const unsigned short* __restrict__ brow0,
                                                   int k0, int half) {
    Frags f;
    // A 16x32 bf16 fragment: lane<16 -> K {k0+0..15}; lane>=16 -> K {k0+16..31}
    f.a.lo = *(const ushort8*)(arow + k0 + half * 16);
    f.a.hi = *(const ushort8*)(arow + k0 + 8 + half * 16);
    // B 32x16 bf16 fragments (contiguous 32B in transposed weights)
    f.b0 = *(const ushort16*)(brow0 + (size_t)(16 * 0) * KCAT + k0);
    f.b1 = *(const ushort16*)(brow0 + (size_t)(16 * 1) * KCAT + k0);
    f.b2 = *(const ushort16*)(brow0 + (size_t)(16 * 2) * KCAT + k0);
    f.b3 = *(const ushort16*)(brow0 + (size_t)(16 * 3) * KCAT + k0);
    return f;
}

// one wave per 16x64 output strip (1 M-tile x 4 N-tiles sharing one A fragment);
// software-pipelined: iteration k+1's fragments load while iteration k's WMMAs run
__global__ void k_gemm_wmma(const unsigned short* __restrict__ Fb,
                            const unsigned short* __restrict__ Wt,
                            const float* __restrict__ bias, float* __restrict__ out,
                            int H, int relu) {
    int wave = blockIdx.x * (blockDim.x >> 5) + (threadIdx.x >> 5);
    int lane = threadIdx.x & 31;
    int nGroups = H >> 6;                  // strips of 64 output columns
    int mTile  = wave / nGroups;
    int nGroup = wave - mTile * nGroups;
    if (mTile >= NN / 16) return;          // uniform across the wave

    int half = lane >> 4;                  // 0: lanes 0-15, 1: lanes 16-31
    int l16  = lane & 15;
    int m = mTile * 16 + l16;              // A row held by this lane
    int nBase = nGroup * 64 + l16;         // first of 4 B/C/D columns for this lane

    v8f c0, c1, c2, c3;
    {
        float b0 = bias[nBase +  0], b1 = bias[nBase + 16];
        float b2 = bias[nBase + 32], b3 = bias[nBase + 48];
#pragma unroll
        for (int r = 0; r < 8; ++r) { c0[r] = b0; c1[r] = b1; c2[r] = b2; c3[r] = b3; }
    }

    const unsigned short* arow  = Fb + (size_t)m * KCAT;
    const unsigned short* brow0 = Wt + (size_t)nBase * KCAT + half * 16;

    Frags cur = load_frags(arow, brow0, 0, half);
#pragma unroll
    for (int k0 = 0; k0 < KCAT; k0 += 32) {
        Frags nxt;
        if (k0 + 32 < KCAT) nxt = load_frags(arow, brow0, k0 + 32, half);
        v16bf av = __builtin_bit_cast(v16bf, cur.a);
        c0 = __builtin_amdgcn_wmma_f32_16x16x32_bf16(false, av, false,
                 __builtin_bit_cast(v16bf, cur.b0), (short)0, c0, false, false);
        c1 = __builtin_amdgcn_wmma_f32_16x16x32_bf16(false, av, false,
                 __builtin_bit_cast(v16bf, cur.b1), (short)0, c1, false, false);
        c2 = __builtin_amdgcn_wmma_f32_16x16x32_bf16(false, av, false,
                 __builtin_bit_cast(v16bf, cur.b2), (short)0, c2, false, false);
        c3 = __builtin_amdgcn_wmma_f32_16x16x32_bf16(false, av, false,
                 __builtin_bit_cast(v16bf, cur.b3), (short)0, c3, false, false);
        if (k0 + 32 < KCAT) cur = nxt;
    }
    // C/D layout: VGPR r, lane half -> row = 16*mTile + r + 8*half
#pragma unroll
    for (int r = 0; r < 8; ++r) {
        int mr = mTile * 16 + r + half * 8;
        float v0 = c0[r], v1 = c1[r], v2 = c2[r], v3 = c3[r];
        if (relu) {
            v0 = v0 > 0.0f ? v0 : 0.0f;
            v1 = v1 > 0.0f ? v1 : 0.0f;
            v2 = v2 > 0.0f ? v2 : 0.0f;
            v3 = v3 > 0.0f ? v3 : 0.0f;
        }
        float* orow = out + (size_t)mr * H + nBase;
        orow[0]  = v0;
        orow[16] = v1;
        orow[32] = v2;
        orow[48] = v3;
    }
}

// ---------------- fused link predictor ----------------
__global__ void k_predictor(const float* __restrict__ h,
                            const int* __restrict__ ps, const int* __restrict__ pd,
                            const int* __restrict__ ns, const int* __restrict__ nd,
                            const float* __restrict__ P1, const float* __restrict__ pb1,
                            const float* __restrict__ P2, const float* __restrict__ pb2,
                            const float* __restrict__ P3, const float* __restrict__ pb3,
                            float* __restrict__ outPos, float* __restrict__ outNeg) {
    __shared__ float sP1[64 * 32], sb1[32], sP2[32 * 16], sb2[16], sP3[16], sb3[1];
    for (int i = threadIdx.x; i < 64 * 32; i += blockDim.x) sP1[i] = P1[i];
    for (int i = threadIdx.x; i < 32 * 16; i += blockDim.x) sP2[i] = P2[i];
    if (threadIdx.x < 32) sb1[threadIdx.x] = pb1[threadIdx.x];
    if (threadIdx.x < 16) { sb2[threadIdx.x] = pb2[threadIdx.x]; sP3[threadIdx.x] = P3[threadIdx.x]; }
    if (threadIdx.x == 0) sb3[0] = pb3[0];
    __syncthreads();

    int i = blockIdx.x * blockDim.x + threadIdx.x;
    if (i >= 2 * PP) return;
    int s, d; float* op;
    if (i < PP) { s = ps[i]; d = pd[i]; op = outPos + i; }
    else        { s = ns[i - PP]; d = nd[i - PP]; op = outNeg + (i - PP); }

    float z[64];
#pragma unroll
    for (int j = 0; j < 64; ++j)
        z[j] = h[(size_t)s * 64 + j] * h[(size_t)d * 64 + j];

    float a1[32];
    for (int o = 0; o < 32; ++o) {
        float acc = sb1[o];
#pragma unroll
        for (int j = 0; j < 64; ++j) acc += z[j] * sP1[j * 32 + o];
        a1[o] = acc > 0.0f ? acc : 0.2f * acc;
    }
    float a2[16];
    for (int o = 0; o < 16; ++o) {
        float acc = sb2[o];
#pragma unroll
        for (int j = 0; j < 32; ++j) acc += a1[j] * sP2[j * 16 + o];
        a2[o] = acc > 0.0f ? acc : 0.2f * acc;
    }
    float acc = sb3[0];
#pragma unroll
    for (int j = 0; j < 16; ++j) acc += a2[j] * sP3[j];
    *op = acc;
}

// ---------------- launcher ----------------
extern "C" void kernel_launch(void* const* d_in, const int* in_sizes, int n_in,
                              void* d_out, int out_size, void* d_ws, size_t ws_size,
                              hipStream_t stream) {
    (void)in_sizes; (void)n_in; (void)out_size; (void)ws_size;

    const float* x      = (const float*)d_in[0];
    const int*   src    = (const int*)  d_in[1];
    const int*   dst    = (const int*)  d_in[2];
    const float* w_edge = (const float*)d_in[3];
    const int*   pos_s  = (const int*)  d_in[4];
    const int*   pos_d  = (const int*)  d_in[5];
    const int*   neg_s  = (const int*)  d_in[6];
    const int*   neg_d  = (const int*)  d_in[7];
    const float* Ws[3]  = { (const float*)d_in[8],  (const float*)d_in[10], (const float*)d_in[12] };
    const float* bs[3]  = { (const float*)d_in[9],  (const float*)d_in[11], (const float*)d_in[13] };
    const float* P1  = (const float*)d_in[14];
    const float* pb1 = (const float*)d_in[15];
    const float* P2  = (const float*)d_in[16];
    const float* pb2 = (const float*)d_in[17];
    const float* P3  = (const float*)d_in[18];
    const float* pb3 = (const float*)d_in[19];

    float* out = (float*)d_out;
    float* h3  = out + 2 * PP;              // final h (N x 64) lives in d_out

    // workspace carve-up (all offsets 32B-aligned)
    float*          ws       = (float*)d_ws;
    float*          deg_norm = ws;                                    // N f32
    float*          F4       = ws + NN;                               // N x 512 f32
    float*          hbuf     = F4 + (size_t)NN * KCAT;                // N x 128 f32
    unsigned short* Fb       = (unsigned short*)(hbuf + (size_t)NN * FIN); // N x 512 bf16
    unsigned short* Wt       = Fb + (size_t)NN * KCAT;                // H x 512 bf16 (max 128x512)

    const int BT = 256;
    const int gN   = (NN + BT - 1) / BT;
    const int gE   = (EE + BT - 1) / BT;
    const int gNF  = (NN * FIN + BT - 1) / BT;
    const int gNK  = (int)(((long long)NN * KCAT + BT - 1) / BT);
    const long long scatterT = (long long)EE * 32;
    const int gScatter = (int)((scatterT + BT - 1) / BT);

    // degrees -> deg_norm
    k_zero<<<gN, BT, 0, stream>>>(deg_norm, NN);
    k_deg_scatter<<<gE, BT, 0, stream>>>(dst, deg_norm);
    k_deg_finalize<<<gN, BT, 0, stream>>>(deg_norm);

    const int   Hs[3]    = { 128, 128, 64 };
    const int   reluF[3] = { 1, 1, 0 };
    const float* layer_in = x;

    for (int l = 0; l < 3; ++l) {
        k_copy_slice0<<<gNF, BT, 0, stream>>>(layer_in, F4);
        for (int hop = 1; hop <= 3; ++hop) {
            float* prevS = F4 + (hop - 1) * FIN;
            float* outS  = F4 + hop * FIN;
            k_zero_slice<<<gNF, BT, 0, stream>>>(outS);
            k_edge_scatter<<<gScatter, BT, 0, stream>>>(src, dst, w_edge, deg_norm, prevS, outS);
            k_scale_slice<<<gNF, BT, 0, stream>>>(outS, deg_norm);
        }
        // one-shot bf16 conversion of operands
        k_cvt_feats<<<gNK, BT, 0, stream>>>(F4, Fb);
        int gW = (KCAT * Hs[l] + BT - 1) / BT;
        k_cvt_wt<<<gW, BT, 0, stream>>>(Ws[l], Wt, Hs[l]);

        float* o = (l < 2) ? hbuf : h3;
        int totalWaves = (NN / 16) * (Hs[l] / 64);   // 1x4-tile strips
        int gGemm = (totalWaves + 7) / 8;            // 8 waves per 256-thread block
        k_gemm_wmma<<<gGemm, BT, 0, stream>>>(Fb, Wt, bs[l], o, Hs[l], reluF[l]);
        layer_in = hbuf;
    }

    int gPred = (2 * PP + BT - 1) / BT;
    k_predictor<<<gPred, BT, 0, stream>>>(h3, pos_s, pos_d, neg_s, neg_d,
                                          P1, pb1, P2, pb2, P3, pb3,
                                          out, out + PP);
}
